// MultiHeadAttention_73100343378638
// MI455X (gfx1250) — matrix operand order repaired and checked by reading, not compiled, not run on hardware
//
#include <hip/hip_runtime.h>
#include <hip/hip_bf16.h>
#include <cstdint>

// ---------------------------------------------------------------------------
// MHA for (N=4, C=256, H=W=48): S=2304, 4 heads, head_dim=64.
// All GEMMs use v_wmma_f32_16x16x32_f16 (f16 operands, f32 accumulate).
// Attention K/V tiles use GLOBAL_LOAD_ASYNC_TO_LDS_B128 (ASYNCcnt) with
// double buffering. Softmax denominator accumulated as a WMMA ones-column.
// ---------------------------------------------------------------------------

typedef __attribute__((ext_vector_type(16))) _Float16 v16h;
typedef __attribute__((ext_vector_type(8)))  _Float16 v8h;
typedef __attribute__((ext_vector_type(8)))  float    v8f;
typedef __attribute__((ext_vector_type(4)))  int      v4i;

#define S_TOT 2304
#define C_TOT 256
#define NH    4
#define HD    64

#if defined(__has_builtin)
#if __has_builtin(__builtin_amdgcn_global_load_async_to_lds_b128) && \
    __has_builtin(__builtin_amdgcn_s_wait_asynccnt)
#define USE_ASYNC 1
#endif
#endif
#ifndef USE_ASYNC
#define USE_ASYNC 0
#endif

// 16-byte global -> LDS copy. Async path goes straight to LDS (no VGPR
// round-trip), tracked by ASYNCcnt; fallback is a plain v8h copy.
__device__ __forceinline__ void cp16_g2l(const _Float16* g, _Float16* l) {
#if USE_ASYNC
  __builtin_amdgcn_global_load_async_to_lds_b128(
      (v4i*)(uintptr_t)g, (v4i*)(uintptr_t)l, 0, 0);
#else
  *(v8h*)l = *(const v8h*)g;
#endif
}

__device__ __forceinline__ void async_wait0() {
#if USE_ASYNC
  __builtin_amdgcn_s_wait_asynccnt(0);
#endif
}

__device__ __forceinline__ v8f wmma16(v16h a, v16h b, v8f c) {
  // (neg_a, A, neg_b, B, c_mod, C, reuse_a, reuse_b)
  return __builtin_amdgcn_wmma_f32_16x16x32_f16(false, a, false, b, (short)0, c,
                                                false, false);
}

// A-operand (16x32 f16, [M][K] row-major tile in LDS): lane l -> M = l&15,
// halves 0..7 = K[8*grp .. +7], halves 8..15 = K[16+8*grp .. +7]
__device__ __forceinline__ v16h make_a(const _Float16* row, int grp) {
  const v8h lo = *(const v8h*)(row + 8 * grp);
  const v8h hi = *(const v8h*)(row + 16 + 8 * grp);
  return __builtin_shufflevector(lo, hi, 0, 1, 2, 3, 4, 5, 6, 7,
                                 8, 9, 10, 11, 12, 13, 14, 15);
}

// B-operand (32x16 f16), tile stored transposed in LDS as [N][K]:
// lane l -> N = l&15, halves 0..7 = K[8*grp .. +7], halves 8..15 = K[16+8*grp .. +7]
__device__ __forceinline__ v16h make_b(const _Float16* rowN, int grp) {
  const v8h lo = *(const v8h*)(rowN + 8 * grp);
  const v8h hi = *(const v8h*)(rowN + 16 + 8 * grp);
  return __builtin_shufflevector(lo, hi, 0, 1, 2, 3, 4, 5, 6, 7,
                                 8, 9, 10, 11, 12, 13, 14, 15);
}

// ---------------------------------------------------------------------------
// Kernel 1: QKV projection.  out[s][o] = (sum_c x[n][c][s]*W[o][c] + b[o])*sc
// MODE 0: store [n][h][S][64] (Q and K); MODE 2: store [n][h][64][S] (V^T)
// sc = 1/sqrt(HD) for Q (folds attention scale), 1.0 otherwise.
// grid (S/64, C/64, N), block 128 (4 waves). Each wave: 16 s-rows x 64 o-cols.
// ---------------------------------------------------------------------------
template <int MODE>
__global__ __launch_bounds__(128) void qkv_proj_kernel(
    const float* __restrict__ x, const float* __restrict__ W,
    const float* __restrict__ bias, _Float16* __restrict__ outbuf, float sc) {
  const int tid = threadIdx.x;
  const int wave = tid >> 5, lane = tid & 31;
  const int grp = lane >> 4, ln = lane & 15;
  const int s0 = blockIdx.x * 64;
  const int o0 = blockIdx.y * 64;
  const int n = blockIdx.z;

  __shared__ __attribute__((aligned(16))) _Float16 lds_a[64 * 40];  // [m][k]
  __shared__ __attribute__((aligned(16))) _Float16 lds_b[64 * 40];  // [o][k]

  v8f acc[4] = {};

  for (int kc = 0; kc < C_TOT; kc += 32) {
    __syncthreads();
    // A tile: x^T (64 s x 32 c), coalesced over s
    for (int idx = tid; idx < 64 * 32; idx += 128) {
      int m = idx & 63, k = idx >> 6;
      lds_a[m * 40 + k] =
          (_Float16)x[(size_t)(n * C_TOT + kc + k) * S_TOT + s0 + m];
    }
    // B tile: W rows (64 o x 32 c), coalesced over c
    for (int idx = tid; idx < 64 * 32; idx += 128) {
      int k = idx & 31, nn = idx >> 5;
      lds_b[nn * 40 + k] = (_Float16)W[(size_t)(o0 + nn) * C_TOT + kc + k];
    }
    __syncthreads();

    v16h a = make_a(&lds_a[(16 * wave + ln) * 40], grp);
#pragma unroll
    for (int ot = 0; ot < 4; ++ot) {
      v16h b = make_b(&lds_b[(16 * ot + ln) * 40], grp);
      acc[ot] = wmma16(a, b, acc[ot]);
    }
  }

  const int sb = s0 + 16 * wave + 8 * grp;
#pragma unroll
  for (int ot = 0; ot < 4; ++ot) {
    int o = o0 + 16 * ot + ln;
    int h = o >> 6, d = o & 63;
    float bv = bias[o];
    if (MODE == 2) {
      // V transposed [n][h][d][S]: 8 consecutive s per lane -> one 16B store
      v8h pk;
#pragma unroll
      for (int r = 0; r < 8; ++r) pk[r] = (_Float16)((acc[ot][r] + bv) * sc);
      *(v8h*)&outbuf[((size_t)(n * NH + h) * HD + d) * S_TOT + sb] = pk;
    } else {
      // Q/K: [n][h][S][64]
#pragma unroll
      for (int r = 0; r < 8; ++r)
        outbuf[((size_t)(n * NH + h) * S_TOT + sb + r) * HD + d] =
            (_Float16)((acc[ot][r] + bv) * sc);
    }
  }
}

// ---------------------------------------------------------------------------
// Kernel 2: flash attention per (n,h). grid (S/64, N*NH), block 128.
// Each wave owns 16 query rows; keys consumed in chunks of 64 with
// double-buffered async global->LDS staging of K and V tiles.
// Softmax denominator L accumulated as WMMA column: L += P @ ones.
// Output O stored f16 as [n][S][256] (heads folded into channel dim).
// ---------------------------------------------------------------------------
__global__ __launch_bounds__(128) void attn_kernel(
    const _Float16* __restrict__ Qb, const _Float16* __restrict__ Kb,
    const _Float16* __restrict__ Vb, _Float16* __restrict__ Ob) {
  const int tid = threadIdx.x;
  const int wave = tid >> 5, lane = tid & 31;
  const int grp = lane >> 4, ln = lane & 15;
  const int q0 = blockIdx.x * 64;
  const int nh = blockIdx.y;
  const int n = nh >> 2, h = nh & 3;

  const _Float16* Q = Qb + (size_t)nh * S_TOT * HD;  // [s][d] (pre-scaled)
  const _Float16* K = Kb + (size_t)nh * S_TOT * HD;  // [s][d]
  const _Float16* V = Vb + (size_t)nh * HD * S_TOT;  // [d][s]

  __shared__ __attribute__((aligned(16))) _Float16 lds_q[64 * 72];     // [q][d]
  __shared__ __attribute__((aligned(16))) _Float16 lds_k[2][64 * 72];  // [key][d]
  __shared__ __attribute__((aligned(16))) _Float16 lds_v[2][64 * 72];  // [d][key]
  __shared__ __attribute__((aligned(16))) _Float16 lds_p[4][16 * 72];  // per-wave P

  // Issue async staging of one 64-key K/V chunk into buffer `buf`.
  auto issue_chunk = [&](int kb, int buf) {
    for (int idx = tid; idx < 64 * 8; idx += 128) {  // K: 64 keys x 64 d
      int row = idx >> 3, ch = idx & 7;
      cp16_g2l(&K[(size_t)(kb + row) * HD + ch * 8],
               &lds_k[buf][row * 72 + ch * 8]);
    }
    for (int idx = tid; idx < 64 * 8; idx += 128) {  // V: 64 d x 64 keys
      int row = idx >> 3, ch = idx & 7;
      cp16_g2l(&V[(size_t)row * S_TOT + kb + ch * 8],
               &lds_v[buf][row * 72 + ch * 8]);
    }
  };

  // Stage Q block (64x64) + first K/V chunk.
  for (int idx = tid; idx < 64 * 8; idx += 128) {
    int row = idx >> 3, ch = idx & 7;
    cp16_g2l(&Q[(size_t)(q0 + row) * HD + ch * 8], &lds_q[row * 72 + ch * 8]);
  }
  issue_chunk(0, 0);
  async_wait0();
  __syncthreads();

  const int qrow = 16 * wave + ln;
  v16h aq0 = make_a(&lds_q[qrow * 72 + 0], grp);
  v16h aq1 = make_a(&lds_q[qrow * 72 + 32], grp);

  // Constant B-operand with ones in column N==0 (rowsum extractor).
  v16h bones;
  {
    _Float16 ov = (ln == 0) ? (_Float16)1.0f : (_Float16)0.0f;
#pragma unroll
    for (int i = 0; i < 16; ++i) bones[i] = ov;
  }

  float rM[8];
#pragma unroll
  for (int r = 0; r < 8; ++r) rM[r] = -1.0e30f;
  v8f oacc[4] = {};
  v8f lacc = {};

  const int nchunks = S_TOT / 64;  // 36

  for (int i = 0; i < nchunks; ++i) {
    const int buf = i & 1;
    if (i) {
      async_wait0();   // my chunk-i loads landed (overlapped with compute i-1)
      __syncthreads(); // all waves' loads visible; buffer buf^1 free to refill
    }
    if (i + 1 < nchunks) issue_chunk((i + 1) * 64, buf ^ 1);

    const _Float16* lk = lds_k[buf];
    const _Float16* lv = lds_v[buf];

    // scores: 4 key-tiles x (2 d-chunks of K=32); Q already carries 1/sqrt(d)
    v8f st[4];
#pragma unroll
    for (int t = 0; t < 4; ++t) {
      v8f s = {};
      v16h b = make_b(&lk[(16 * t + ln) * 72 + 0], grp);
      s = wmma16(aq0, b, s);
      b = make_b(&lk[(16 * t + ln) * 72 + 32], grp);
      st[t] = wmma16(aq1, b, s);
    }

    // online softmax update (row = r + 8*grp; 16-lane half-wave max)
#pragma unroll
    for (int r = 0; r < 8; ++r) {
      float a0 = st[0][r], a1 = st[1][r], a2 = st[2][r], a3 = st[3][r];
      float mx = fmaxf(fmaxf(a0, a1), fmaxf(a2, a3));
      mx = fmaxf(mx, __shfl_xor(mx, 1));
      mx = fmaxf(mx, __shfl_xor(mx, 2));
      mx = fmaxf(mx, __shfl_xor(mx, 4));
      mx = fmaxf(mx, __shfl_xor(mx, 8));
      float newM = fmaxf(rM[r], mx);
      float corr = __expf(rM[r] - newM);
      rM[r] = newM;
      lacc[r] *= corr;
#pragma unroll
      for (int dt = 0; dt < 4; ++dt) oacc[dt][r] *= corr;
      // scatter P into per-wave A-layout tile [m][k], k = 16*t + ln
      int m = r + 8 * grp;
      lds_p[wave][m * 72 + ln +  0] = (_Float16)__expf(a0 - newM);
      lds_p[wave][m * 72 + ln + 16] = (_Float16)__expf(a1 - newM);
      lds_p[wave][m * 72 + ln + 32] = (_Float16)__expf(a2 - newM);
      lds_p[wave][m * 72 + ln + 48] = (_Float16)__expf(a3 - newM);
    }

    // O += P @ V_chunk; L += P @ ones  (2 key sub-chunks of K=32)
#pragma unroll
    for (int j = 0; j < 2; ++j) {
      v16h ap = make_a(&lds_p[wave][ln * 72 + 32 * j], grp);
#pragma unroll
      for (int dt = 0; dt < 4; ++dt) {
        v16h bv = make_b(&lv[(16 * dt + ln) * 72 + 32 * j], grp);
        oacc[dt] = wmma16(ap, bv, oacc[dt]);
      }
      lacc = wmma16(ap, bones, lacc);
    }
  }

  // broadcast L (column N==0 lives in lane 0 / lane 16), normalize, store
  float L[8];
#pragma unroll
  for (int r = 0; r < 8; ++r) L[r] = __shfl(lacc[r], lane & 16);

  _Float16* Op = Ob + (size_t)n * S_TOT * C_TOT + h * HD;
#pragma unroll
  for (int dt = 0; dt < 4; ++dt) {
    int d = 16 * dt + ln;
#pragma unroll
    for (int r = 0; r < 8; ++r) {
      int s = q0 + 16 * wave + 8 * grp + r;
      Op[(size_t)s * C_TOT + d] = (_Float16)(oacc[dt][r] / L[r]);
    }
  }
}

// ---------------------------------------------------------------------------
// Kernel 3: output projection + bias + residual.
// out[n][o][s] = x[n][o][s] + bo[o] + sum_c O[n][s][c] * Wo[o][c]
// ---------------------------------------------------------------------------
__global__ __launch_bounds__(128) void out_proj_kernel(
    const _Float16* __restrict__ Ob, const float* __restrict__ Wo,
    const float* __restrict__ bo, const float* __restrict__ x,
    float* __restrict__ out) {
  const int tid = threadIdx.x;
  const int wave = tid >> 5, lane = tid & 31;
  const int grp = lane >> 4, ln = lane & 15;
  const int s0 = blockIdx.x * 64;
  const int o0 = blockIdx.y * 64;
  const int n = blockIdx.z;

  __shared__ __attribute__((aligned(16))) _Float16 lds_a[64 * 40];
  __shared__ __attribute__((aligned(16))) _Float16 lds_b[64 * 40];

  v8f acc[4] = {};

  for (int kc = 0; kc < C_TOT; kc += 32) {
    __syncthreads();
    // A: O rows (64 s x 32 c), already f16 row-major -> 16B copies
    for (int idx = tid; idx < 64 * 4; idx += 128) {
      int m = idx >> 2, ch = idx & 3;
      *(v8h*)&lds_a[m * 40 + ch * 8] =
          *(const v8h*)&Ob[((size_t)(n * S_TOT + s0 + m)) * C_TOT + kc + ch * 8];
    }
    // B: Wo tile
    for (int idx = tid; idx < 64 * 32; idx += 128) {
      int k = idx & 31, nn = idx >> 5;
      lds_b[nn * 40 + k] = (_Float16)Wo[(size_t)(o0 + nn) * C_TOT + kc + k];
    }
    __syncthreads();

    v16h a = make_a(&lds_a[(16 * wave + ln) * 40], grp);
#pragma unroll
    for (int ot = 0; ot < 4; ++ot) {
      v16h b = make_b(&lds_b[(16 * ot + ln) * 40], grp);
      acc[ot] = wmma16(a, b, acc[ot]);
    }
  }

  const int sb = s0 + 16 * wave + 8 * grp;  // 8 consecutive s per lane
#pragma unroll
  for (int ot = 0; ot < 4; ++ot) {
    int o = o0 + 16 * ot + ln;
    float bv = bo[o];
    size_t gi = (size_t)(n * C_TOT + o) * S_TOT + sb;
    float4 a0 = *(const float4*)&x[gi];
    float4 a1 = *(const float4*)&x[gi + 4];
    a0.x += bv + acc[ot][0];
    a0.y += bv + acc[ot][1];
    a0.z += bv + acc[ot][2];
    a0.w += bv + acc[ot][3];
    a1.x += bv + acc[ot][4];
    a1.y += bv + acc[ot][5];
    a1.z += bv + acc[ot][6];
    a1.w += bv + acc[ot][7];
    *(float4*)&out[gi] = a0;
    *(float4*)&out[gi + 4] = a1;
  }
}

// ---------------------------------------------------------------------------
extern "C" void kernel_launch(void* const* d_in, const int* in_sizes, int n_in,
                              void* d_out, int out_size, void* d_ws,
                              size_t ws_size, hipStream_t stream) {
  const float* x  = (const float*)d_in[0];
  const float* Wq = (const float*)d_in[1];
  const float* bq = (const float*)d_in[2];
  const float* Wk = (const float*)d_in[3];
  const float* bk = (const float*)d_in[4];
  const float* Wv = (const float*)d_in[5];
  const float* bv = (const float*)d_in[6];
  const float* Wo = (const float*)d_in[7];
  const float* bo = (const float*)d_in[8];
  float* out = (float*)d_out;

  const size_t elems = (size_t)4 * NH * S_TOT * HD;  // 2,359,296 per buffer
  _Float16* Qb = (_Float16*)d_ws;
  _Float16* Kb = Qb + elems;
  _Float16* Vb = Kb + elems;
  _Float16* Ob = Vb + elems;  // [n][S][256], same element count

  dim3 blk(128);
  dim3 g1(S_TOT / 64, C_TOT / 64, 4);
  qkv_proj_kernel<0><<<g1, blk, 0, stream>>>(x, Wq, bq, Qb, 0.125f);  // Q*1/sqrt(d)
  qkv_proj_kernel<0><<<g1, blk, 0, stream>>>(x, Wk, bk, Kb, 1.0f);
  qkv_proj_kernel<2><<<g1, blk, 0, stream>>>(x, Wv, bv, Vb, 1.0f);

  dim3 g2(S_TOT / 64, 4 * NH);
  attn_kernel<<<g2, blk, 0, stream>>>(Qb, Kb, Vb, Ob);

  out_proj_kernel<<<g1, blk, 0, stream>>>(Ob, Wo, bo, x, out);
}
